// TianCLDNN_10204842295641
// MI455X (gfx1250) — matrix operand order repaired and checked by reading, not compiled, not run on hardware
//
#include <hip/hip_runtime.h>
#include <hip/hip_bf16.h>

// CLDNN pipeline for MI455X (gfx1250, wave32, WMMA).
//
// Stages:
//   prep:   convert weights to f16 (padded/reordered K layouts), zero flat buffer
//   conv:   one wave32 per (b,t) sample; conv1 in VALU -> LDS(f16, transposed);
//           conv2/3/4 as WMMA GEMMs (v_wmma_f32_16x16x32_f16) with BN fused;
//           all WMMA fragments loaded as b128 vectors (no scalar gathers)
//   linear: 8192x80 @ 80x16 via WMMA, ReLU
//   gi:     GRU input projections (K=16/20 -> N=30) via WMMA
//   scan:   sequential GRU recurrences, 64 single-wave WGs (batch x direction)
//   out:    20 -> 1 projection

typedef __attribute__((ext_vector_type(16))) _Float16 v16h;
typedef __attribute__((ext_vector_type(8)))  _Float16 v8h;
typedef __attribute__((ext_vector_type(8)))  float    v8f;
typedef __attribute__((ext_vector_type(4)))  float    v4f;

#define NB   32     // batch
#define NT   256    // time steps
#define NW   48     // features
#define NL   29     // window length (LEFT+RIGHT)
#define NMT  227    // valid time positions
#define NH   10     // GRU hidden

#define X1S  264    // x1T row stride (halfs): 528B rows -> conflict-free b128 lanes
#define X2S  24     // x2T row stride (halfs): 48B rows
#define X3S  24     // x3T row stride (halfs)

__device__ __forceinline__ float sigmoidf_(float x) {
  return 1.0f / (1.0f + __expf(-x));
}

__device__ __forceinline__ v16h cat8(v8h a, v8h b) {
  return __builtin_shufflevector(a, b, 0, 1, 2, 3, 4, 5, 6, 7,
                                 8, 9, 10, 11, 12, 13, 14, 15);
}

// A fragment (16x32 f16): two contiguous 8-half runs per lane.
// lane<16: K in {base..base+7, base+16..base+23}; lane>=16: +8.
__device__ __forceinline__ v16h load_a_frag(const _Float16* row, int kbase, int hi) {
  v8h a0 = *(const v8h*)&row[kbase + hi * 8];
  v8h a1 = *(const v8h*)&row[kbase + 16 + hi * 8];
  return cat8(a0, a1);
}

// B fragment (32x16 f16) from K-contiguous storage: 16 contiguous halfs per lane.
__device__ __forceinline__ v16h load_b_frag(const _Float16* p) {
  v8h b0 = *(const v8h*)&p[0];
  v8h b1 = *(const v8h*)&p[8];
  return cat8(b0, b1);
}

// ---------------------------------------------------------------- prep kernels

__global__ void zero_f32_kernel(float* p, int n) {
  int i = blockIdx.x * blockDim.x + threadIdx.x;
  if (i < n) p[i] = 0.0f;
}

// dst[m*Kd + k] = (m<Mv && k<Ks) ? f16(src[m*Ks + k]) : 0
__global__ void cvt_plain_kernel(_Float16* dst, const float* src,
                                 int Mv, int Ks, int Kd, int Mtot) {
  int i = blockIdx.x * blockDim.x + threadIdx.x;
  if (i >= Mtot * Kd) return;
  int m = i / Kd, k = i - m * Kd;
  dst[i] = (m < Mv && k < Ks) ? (_Float16)src[m * Ks + k] : (_Float16)0.0f;
}

// conv3/conv4 weights: src[co][ci][kw] -> dst[co][k] with k = kw*16 + ci
__global__ void cvt_ckw_kernel(_Float16* dst, const float* src,
                               int Cin, int Kw, int Kd) {
  int i = blockIdx.x * blockDim.x + threadIdx.x;
  if (i >= 16 * Kd) return;
  int m = i / Kd, k = i - m * Kd;
  int kw = k >> 4, ci = k & 15;
  dst[i] = (kw < Kw) ? (_Float16)src[(m * Cin + ci) * Kw + kw] : (_Float16)0.0f;
}

// ---------------------------------------------------------------- conv stack

struct ConvParams {
  const float* seq;
  const float* w1;
  const _Float16* w2h;   // [16][256], k = ci*8+kh
  const _Float16* w3h;   // [16][160], k = kw*16+ci (kw<9), rest 0
  const _Float16* w4h;   // [16][96],  k = kw*16+ci (kw<6)
  const float *g2, *b2, *m2, *v2;
  const float *g3, *b3, *m3, *v3;
  const float *g4, *b4, *m4, *v4;
  float* flat;           // [B][256][80], pad rows pre-zeroed
};

__global__ __launch_bounds__(32) void conv_stack_kernel(ConvParams p) {
  __shared__ float    winS[NL * NW];        // 29x48 input window
  __shared__ _Float16 x1T[NW * X1S];        // conv1 out, transposed [w][k=ci*8+kh]
  __shared__ _Float16 x2T[NW * X2S];        // conv2 out (BN+ReLU), [w][ci]
  __shared__ _Float16 x3T[14 * X3S];        // conv3 out (BN), [w'][ci]

  const int s = blockIdx.x;
  const int b = s / NMT;
  const int t = s - b * NMT;
  const int lane = threadIdx.x;
  const int hi = lane >> 4;       // 0 / 1
  const int lo = lane & 15;

  __builtin_prefetch(&p.seq[(b * NT + t) * NW], 0, 3);   // global_prefetch_b8

  // stage window into LDS
  for (int i = lane; i < NL * NW; i += 32) {
    int hh = i / NW, ww = i - hh * NW;
    winS[i] = p.seq[(b * NT + t + hh) * NW + ww];
  }
  // this lane's conv1 channel taps stay in registers (ci == lane)
  float w1r[8];
#pragma unroll
  for (int j = 0; j < 8; j++) w1r[j] = p.w1[lane * 8 + j];
  __syncthreads();

  // conv1 + ReLU, stored transposed: x1T[w][ci*8+kh]
  for (int w = 0; w < NW; w++) {
    float col[NL];
#pragma unroll
    for (int r = 0; r < NL; r++) col[r] = winS[r * NW + w];
    v8h pack;
#pragma unroll
    for (int kh = 0; kh < 8; kh++) {
      float a = 0.0f;
#pragma unroll
      for (int j = 0; j < 8; j++) a += col[kh * 3 + j] * w1r[j];
      pack[kh] = (_Float16)fmaxf(a, 0.0f);
    }
    *(v8h*)&x1T[w * X1S + lane * 8] = pack;   // one ds_store_b128
  }
  __syncthreads();

  // conv2: D[co][w] = sum_{k<256} w2[co][k] * x1[k][w], three 16-wide N tiles
  float inv2[8], add2[8];
#pragma unroll
  for (int j = 0; j < 8; j++) {
    int c = hi * 8 + j;
    float iv = p.g2[c] * rsqrtf(p.v2[c] + 1e-5f);
    inv2[j] = iv;
    add2[j] = p.b2[c] - p.m2[c] * iv;
  }
  for (int nt = 0; nt < 3; nt++) {
    v8f acc = {};
    for (int kb = 0; kb < 8; kb++) {
      v16h A = load_a_frag(&p.w2h[lo * 256], kb * 32, hi);
      v16h Bm = load_a_frag(&x1T[(nt * 16 + lo) * X1S], kb * 32, hi);
      acc = __builtin_amdgcn_wmma_f32_16x16x32_f16(false, A, false, Bm,
                                                   (short)0, acc, false, false);
    }
    v8h pack;
#pragma unroll
    for (int j = 0; j < 8; j++) {
      float y = acc[j] * inv2[j] + add2[j];             // BN2
      pack[j] = (_Float16)fmaxf(y, 0.0f);
    }
    *(v8h*)&x2T[(nt * 16 + lo) * X2S + hi * 8] = pack;  // [w][ci], b128 store
  }
  __syncthreads();

  // conv3: D[co][w'] = sum_{ci,kw<9} w3[co][ci][kw] * x2[ci][3w'+kw], w'<14
  float inv3[8], add3[8];
#pragma unroll
  for (int j = 0; j < 8; j++) {
    int c = hi * 8 + j;
    float iv = p.g3[c] * rsqrtf(p.v3[c] + 1e-5f);
    inv3[j] = iv;
    add3[j] = p.b3[c] - p.m3[c] * iv;
  }
  {
    v8f acc = {};
    for (int kb = 0; kb < 5; kb++) {
      v16h A = load_b_frag(&p.w3h[lo * 160 + kb * 32 + hi * 16]);
      int kw = kb * 2 + hi;                 // fragment-uniform: K base is 16-aligned
      v16h Bm = {};
      if (lo < 14 && kw < 9)
        Bm = load_b_frag(&x2T[(3 * lo + kw) * X2S]);   // ci = j, contiguous
      acc = __builtin_amdgcn_wmma_f32_16x16x32_f16(false, A, false, Bm,
                                                   (short)0, acc, false, false);
    }
    if (lo < 14) {
      v8h pack;
#pragma unroll
      for (int j = 0; j < 8; j++) {
        float y = acc[j] * inv3[j] + add3[j];           // BN3, no ReLU
        pack[j] = (_Float16)y;
      }
      *(v8h*)&x3T[lo * X3S + hi * 8] = pack;
    }
  }
  __syncthreads();

  // conv4: D[co][w''] = sum_{ci,kw<6} w4[co][ci][kw] * x3[ci][2w''+kw], w''<5
  float inv4[8], add4[8];
#pragma unroll
  for (int j = 0; j < 8; j++) {
    int c = hi * 8 + j;
    float iv = p.g4[c] * rsqrtf(p.v4[c] + 1e-5f);
    inv4[j] = iv;
    add4[j] = p.b4[c] - p.m4[c] * iv;
  }
  {
    v8f acc = {};
    for (int kb = 0; kb < 3; kb++) {
      v16h A = load_b_frag(&p.w4h[lo * 96 + kb * 32 + hi * 16]);
      int kw = kb * 2 + hi;                 // < 6 always
      v16h Bm = {};
      if (lo < 5)
        Bm = load_b_frag(&x3T[(2 * lo + kw) * X3S]);
      acc = __builtin_amdgcn_wmma_f32_16x16x32_f16(false, A, false, Bm,
                                                   (short)0, acc, false, false);
    }
    if (lo < 5) {
#pragma unroll
      for (int j = 0; j < 8; j++) {
        int c = hi * 8 + j;
        float y = acc[j] * inv4[j] + add4[j];           // BN4
        // flat[b][t+LEFT][c*5 + w'']  (LEFT = 14)
        p.flat[(b * NT + t + 14) * 80 + c * 5 + lo] = y;
      }
    }
  }
}

// ---------------------------------------------------------------- linear 80->16

__global__ __launch_bounds__(32) void linear_kernel(const float* __restrict__ X,
                                                    const _Float16* __restrict__ W,
                                                    const float* __restrict__ bias,
                                                    float* __restrict__ Y) {
  const int tile = blockIdx.x;            // 512 tiles of 16 rows
  const int lane = threadIdx.x;
  const int hi = lane >> 4, lo = lane & 15;
  const float* xrow = &X[(tile * 16 + lo) * 80];
  v8f acc = {};
  for (int kb = 0; kb < 3; kb++) {        // K = 80 padded to 96
    // A: two 8-float contiguous runs, f32 -> f16 (runs are 8-aligned vs K=80)
    v8h a01[2];
#pragma unroll
    for (int g = 0; g < 2; g++) {
      int base = kb * 32 + g * 16 + hi * 8;
      v8h r = {};
      if (base < 80) {
        v4f f0 = *(const v4f*)&xrow[base];
        v4f f1 = *(const v4f*)&xrow[base + 4];
#pragma unroll
        for (int j = 0; j < 4; j++) { r[j] = (_Float16)f0[j]; r[4 + j] = (_Float16)f1[j]; }
      }
      a01[g] = r;
    }
    v16h A = cat8(a01[0], a01[1]);
    v16h Bm = load_a_frag(&W[lo * 96], kb * 32, hi);  // B[k][n] = lin_w[n][k]
    acc = __builtin_amdgcn_wmma_f32_16x16x32_f16(false, A, false, Bm,
                                                 (short)0, acc, false, false);
  }
  float bb = bias[lo];
#pragma unroll
  for (int j = 0; j < 8; j++) {
    int r = tile * 16 + hi * 8 + j;
    Y[r * 16 + lo] = fmaxf(acc[j] + bb, 0.0f);
  }
}

// ---------------------------------------------------------------- GRU input proj

// gi[r][n] = sum_k X[r][k] * Wih[n][k] + bih[n],  n < 30, stored in [8192][32]
__global__ __launch_bounds__(32) void gi_kernel(const float* __restrict__ X,
                                                int ldx, int Kin,
                                                const _Float16* __restrict__ W,
                                                const float* __restrict__ bih,
                                                float* __restrict__ gi) {
  const int tile = blockIdx.x;
  const int lane = threadIdx.x;
  const int hi = lane >> 4, lo = lane & 15;
  v16h A;
#pragma unroll
  for (int j = 0; j < 16; j++) {
    int kk = ((j >> 3) << 4) + (hi << 3) + (j & 7);   // single K block (Kpad=32)
    A[j] = (kk < Kin) ? (_Float16)X[(tile * 16 + lo) * ldx + kk] : (_Float16)0.0f;
  }
  for (int nt = 0; nt < 2; nt++) {
    v16h Bm = load_a_frag(&W[(nt * 16 + lo) * 32], 0, hi);  // rows 30,31 zero
    v8f acc = {};
    acc = __builtin_amdgcn_wmma_f32_16x16x32_f16(false, A, false, Bm,
                                                 (short)0, acc, false, false);
    int n = nt * 16 + lo;
    float bb = (n < 30) ? bih[n] : 0.0f;
#pragma unroll
    for (int j = 0; j < 8; j++) {
      int r = tile * 16 + hi * 8 + j;
      gi[r * 32 + n] = acc[j] + bb;
    }
  }
}

// ---------------------------------------------------------------- GRU scan

struct ScanParams {
  const float* gi[2];     // [8192][32], fwd / bwd
  const float* Whh[2];    // [30][10]
  const float* bhh[2];    // [30]
  float* hout;            // [8192][20]; dir writes cols dir*10 .. dir*10+9
};

__global__ __launch_bounds__(32) void gru_scan_kernel(ScanParams sp) {
  __shared__ float WhS[300];
  __shared__ float bhS[30];
  __shared__ float hS[10];
  __shared__ float ghS[30];

  const int b = blockIdx.x >> 1;
  const int dir = blockIdx.x & 1;
  const int lane = threadIdx.x;
  const float* gi  = sp.gi[dir];
  const float* Whh = sp.Whh[dir];
  const float* bhh = sp.bhh[dir];

  for (int i = lane; i < 300; i += 32) WhS[i] = Whh[i];
  if (lane < 30) bhS[lane] = bhh[lane];
  if (lane < 10) hS[lane] = 0.0f;
  __syncthreads();

  for (int s = 0; s < NT; s++) {
    int t = dir ? (NT - 1 - s) : s;
    int r = b * NT + t;
    if (lane < 30) {
      float a = bhS[lane];
#pragma unroll
      for (int i = 0; i < 10; i++) a += WhS[lane * 10 + i] * hS[i];
      ghS[lane] = a;
    }
    __syncthreads();
    if (lane < 10) {
      float gr = gi[r * 32 + lane];
      float gz = gi[r * 32 + 10 + lane];
      float gn = gi[r * 32 + 20 + lane];
      float rr = sigmoidf_(gr + ghS[lane]);
      float zz = sigmoidf_(gz + ghS[10 + lane]);
      float nn = tanhf(gn + rr * ghS[20 + lane]);
      float hn = (1.0f - zz) * nn + zz * hS[lane];
      sp.hout[r * 20 + dir * 10 + lane] = hn;
      hS[lane] = hn;
    }
    __syncthreads();
  }
}

// ---------------------------------------------------------------- output proj

__global__ void out_proj_kernel(const float* __restrict__ H,
                                const float* __restrict__ out_w,
                                const float* __restrict__ out_b,
                                float* __restrict__ out) {
  int r = blockIdx.x * blockDim.x + threadIdx.x;
  if (r >= NB * NT) return;
  float a = out_b[0];
#pragma unroll
  for (int c = 0; c < 20; c++) a += H[r * 20 + c] * out_w[c];
  out[r] = a;
}

// ---------------------------------------------------------------- launch

extern "C" void kernel_launch(void* const* d_in, const int* in_sizes, int n_in,
                              void* d_out, int out_size, void* d_ws, size_t ws_size,
                              hipStream_t stream) {
  (void)in_sizes; (void)n_in; (void)out_size; (void)ws_size;
  // pytree-leaf order of setup_inputs():
  const float* seq   = (const float*)d_in[0];
  const float* w1    = (const float*)d_in[1];
  const float* w2    = (const float*)d_in[2];
  const float* w3    = (const float*)d_in[3];
  const float* w4    = (const float*)d_in[4];
  const float* bn2g  = (const float*)d_in[5];
  const float* bn2b  = (const float*)d_in[6];
  const float* bn2m  = (const float*)d_in[7];
  const float* bn2v  = (const float*)d_in[8];
  const float* bn3g  = (const float*)d_in[9];
  const float* bn3b  = (const float*)d_in[10];
  const float* bn3m  = (const float*)d_in[11];
  const float* bn3v  = (const float*)d_in[12];
  const float* bn4g  = (const float*)d_in[13];
  const float* bn4b  = (const float*)d_in[14];
  const float* bn4m  = (const float*)d_in[15];
  const float* bn4v  = (const float*)d_in[16];
  const float* lin_w = (const float*)d_in[17];
  const float* lin_b = (const float*)d_in[18];
  // GRU leaves: layer1 fwd(19-22) bwd(23-26), layer2 fwd(27-30) bwd(31-34)
  const float* wih[4]; const float* whh[4]; const float* bih[4]; const float* bhh[4];
  for (int d = 0; d < 4; d++) {
    wih[d] = (const float*)d_in[19 + 4 * d + 0];
    whh[d] = (const float*)d_in[19 + 4 * d + 1];
    bih[d] = (const float*)d_in[19 + 4 * d + 2];
    bhh[d] = (const float*)d_in[19 + 4 * d + 3];
  }
  const float* out_w = (const float*)d_in[35];
  const float* out_b = (const float*)d_in[36];
  float* out = (float*)d_out;

  // workspace carve-out
  char* ws = (char*)d_ws;
  size_t off = 0;
  auto alloc = [&](size_t bytes) -> void* {
    void* p = ws + off;
    off = (off + bytes + 255) & ~(size_t)255;
    return p;
  };
  float*     flat   = (float*)alloc((size_t)NB * NT * 80 * 4);
  float*     h0     = (float*)alloc((size_t)NB * NT * 16 * 4);
  float*     h1     = (float*)alloc((size_t)NB * NT * 20 * 4);
  float*     h2     = (float*)alloc((size_t)NB * NT * 20 * 4);
  float*     gif    = (float*)alloc((size_t)NB * NT * 32 * 4);
  float*     gib    = (float*)alloc((size_t)NB * NT * 32 * 4);
  _Float16*  w2h    = (_Float16*)alloc(16 * 256 * 2);
  _Float16*  w3h    = (_Float16*)alloc(16 * 160 * 2);
  _Float16*  w4h    = (_Float16*)alloc(16 * 96 * 2);
  _Float16*  linw16 = (_Float16*)alloc(16 * 96 * 2);
  _Float16*  wih16[4];
  for (int d = 0; d < 4; d++) wih16[d] = (_Float16*)alloc(32 * 32 * 2);

  // ---- prep
  {
    int n = NB * NT * 80;
    zero_f32_kernel<<<(n + 255) / 256, 256, 0, stream>>>(flat, n);
  }
  cvt_plain_kernel<<<(16 * 256 + 255) / 256, 256, 0, stream>>>(w2h, w2, 16, 256, 256, 16);
  cvt_ckw_kernel<<<(16 * 160 + 255) / 256, 256, 0, stream>>>(w3h, w3, 16, 9, 160);
  cvt_ckw_kernel<<<(16 * 96 + 255) / 256, 256, 0, stream>>>(w4h, w4, 16, 6, 96);
  cvt_plain_kernel<<<(16 * 96 + 255) / 256, 256, 0, stream>>>(linw16, lin_w, 16, 80, 96, 16);
  cvt_plain_kernel<<<(32 * 32 + 255) / 256, 256, 0, stream>>>(wih16[0], wih[0], 30, 16, 32, 32);
  cvt_plain_kernel<<<(32 * 32 + 255) / 256, 256, 0, stream>>>(wih16[1], wih[1], 30, 16, 32, 32);
  cvt_plain_kernel<<<(32 * 32 + 255) / 256, 256, 0, stream>>>(wih16[2], wih[2], 30, 20, 32, 32);
  cvt_plain_kernel<<<(32 * 32 + 255) / 256, 256, 0, stream>>>(wih16[3], wih[3], 30, 20, 32, 32);

  // ---- conv stack: one wave32 per sample
  ConvParams cp;
  cp.seq = seq; cp.w1 = w1; cp.w2h = w2h; cp.w3h = w3h; cp.w4h = w4h;
  cp.g2 = bn2g; cp.b2 = bn2b; cp.m2 = bn2m; cp.v2 = bn2v;
  cp.g3 = bn3g; cp.b3 = bn3b; cp.m3 = bn3m; cp.v3 = bn3v;
  cp.g4 = bn4g; cp.b4 = bn4b; cp.m4 = bn4m; cp.v4 = bn4v;
  cp.flat = flat;
  conv_stack_kernel<<<NB * NMT, 32, 0, stream>>>(cp);

  // ---- linear 80 -> 16 (+ReLU), 512 row tiles
  linear_kernel<<<512, 32, 0, stream>>>(flat, linw16, lin_b, h0);

  // ---- GRU layer 1
  gi_kernel<<<512, 32, 0, stream>>>(h0, 16, 16, wih16[0], bih[0], gif);
  gi_kernel<<<512, 32, 0, stream>>>(h0, 16, 16, wih16[1], bih[1], gib);
  {
    ScanParams sp;
    sp.gi[0] = gif;    sp.gi[1] = gib;
    sp.Whh[0] = whh[0]; sp.Whh[1] = whh[1];
    sp.bhh[0] = bhh[0]; sp.bhh[1] = bhh[1];
    sp.hout = h1;
    gru_scan_kernel<<<NB * 2, 32, 0, stream>>>(sp);
  }

  // ---- GRU layer 2
  gi_kernel<<<512, 32, 0, stream>>>(h1, 20, 20, wih16[2], bih[2], gif);
  gi_kernel<<<512, 32, 0, stream>>>(h1, 20, 20, wih16[3], bih[3], gib);
  {
    ScanParams sp;
    sp.gi[0] = gif;    sp.gi[1] = gib;
    sp.Whh[0] = whh[2]; sp.Whh[1] = whh[3];
    sp.bhh[0] = bhh[2]; sp.bhh[1] = bhh[3];
    sp.hout = h2;
    gru_scan_kernel<<<NB * 2, 32, 0, stream>>>(sp);
  }

  // ---- output projection
  out_proj_kernel<<<(NB * NT + 255) / 256, 256, 0, stream>>>(h2, out_w, out_b, out);
}
